// CLIP_69458211111620
// MI455X (gfx1250) — matrix operand order repaired and checked by reading, not compiled, not run on hardware
//
#include <hip/hip_runtime.h>
#include <hip/hip_bf16.h>
#include <math.h>

// ---------------------------------------------------------------------------
// Types for CDNA5 WMMA (wave32): v16bf A/B fragments, v8f accumulator.
// ---------------------------------------------------------------------------
typedef __bf16 bf16_t;
typedef __attribute__((ext_vector_type(16))) __bf16 v16bf;
typedef __attribute__((ext_vector_type(8)))  float  v8f;

#define TILE_M 128
#define TILE_N 128
#define TILE_K 32
#define GEMM_THREADS 256   // 8 waves (wave32), arranged 2 x 4

union FragA { uint4 u[2]; v16bf v; };

static __device__ __forceinline__ float bf2f(bf16_t b) { return (float)b; }
static __device__ __forceinline__ bf16_t f2bf(float f) { return (bf16_t)f; }

// ---------------------------------------------------------------------------
// gfx1250 async global->LDS copy (GLOBAL_LOAD_ASYNC_TO_LDS_B128, ASYNCcnt).
// Probe-confirmed: builtin takes (v4i addrspace(1)*, v4i addrspace(3)*, ...).
// ---------------------------------------------------------------------------
typedef int v4i_t __attribute__((vector_size(4 * sizeof(int))));

static __device__ __forceinline__ void async_ld_b128(bf16_t* lds_dst,
                                                     const bf16_t* gsrc)
{
#if __has_builtin(__builtin_amdgcn_global_load_async_to_lds_b128)
  typedef __attribute__((address_space(1))) v4i_t* gp_t;
  typedef __attribute__((address_space(3))) v4i_t* sp_t;
  __builtin_amdgcn_global_load_async_to_lds_b128((gp_t)gsrc, (sp_t)lds_dst, 0, 0);
#else
  // VDST = LDS byte offset (low 32 bits of the flat shared address),
  // VADDR = 64-bit global address.
  unsigned off = (unsigned)(uintptr_t)lds_dst;
  asm volatile("global_load_async_to_lds_b128 %0, %1, off"
               :: "v"(off), "v"(gsrc) : "memory");
#endif
}

static __device__ __forceinline__ void wait_async()
{
#if __has_builtin(__builtin_amdgcn_s_wait_asynccnt)
  __builtin_amdgcn_s_wait_asynccnt(0);
#else
  asm volatile("s_wait_asynccnt 0x0" ::: "memory");
#endif
}

// ---------------------------------------------------------------------------
// Async global -> LDS tile staging. A is [M][K] row-major bf16 (lda == K),
// Bt is [N][K] row-major bf16 (ldb == K). K is a multiple of 32; M/N edge
// rows are simply not transferred (LDS pre-zeroed for edge blocks).
// ---------------------------------------------------------------------------
static __device__ __forceinline__ void load_tiles_async(
    bf16_t* __restrict__ dA, bf16_t* __restrict__ dB,
    const bf16_t* __restrict__ Ab, const bf16_t* __restrict__ Bb,
    int bM, int bN, int M, int N, int K, int kt, int tid)
{
  const int k0 = kt * TILE_K;
  const bool pf = (k0 + 3 * TILE_K <= K);
#pragma unroll
  for (int t = 0; t < 2; ++t) {
    int li  = tid + t * GEMM_THREADS;   // 0..511
    int row = li >> 2;                  // 4 x b128 per 32-elem row
    int col = (li & 3) << 3;            // bf16 column
    int gm  = bM + row;
    if (gm < M) {
      const bf16_t* src = Ab + (size_t)gm * K + (k0 + col);
      async_ld_b128(dA + row * TILE_K + col, src);
      if (pf) __builtin_prefetch(src + 2 * TILE_K, 0, 3);
    }
  }
#pragma unroll
  for (int t = 0; t < 2; ++t) {
    int li  = tid + t * GEMM_THREADS;
    int row = li >> 2;
    int col = (li & 3) << 3;
    int gn  = bN + row;
    if (gn < N) {
      const bf16_t* src = Bb + (size_t)gn * K + (k0 + col);
      async_ld_b128(dB + row * TILE_K + col, src);
      if (pf) __builtin_prefetch(src + 2 * TILE_K, 0, 3);
    }
  }
}

// ---------------------------------------------------------------------------
// Generic batched bf16 GEMM: out[b] = act( A[b](MxK) * Bt[b](NxK)^T * alpha
//                                          + bias[n] + *addend )
// Epilogue can emit fp32, bf16 [M][N], and/or bf16 transposed [N][M].
// act: 0 = none, 1 = leaky_relu(0.01), 2 = exact GELU.
// ---------------------------------------------------------------------------
__global__ __launch_bounds__(GEMM_THREADS)
void gemm_bf16_wmma(const bf16_t* __restrict__ A,  size_t strideA,
                    const bf16_t* __restrict__ Bt, size_t strideB,
                    const float* __restrict__ bias,
                    const float* __restrict__ addend,
                    float alpha, int act,
                    float*  __restrict__ outF,
                    bf16_t* __restrict__ outB,
                    bf16_t* __restrict__ outBT,
                    int M, int N, int K)
{
  __shared__ __align__(32) bf16_t lsA[2][TILE_M * TILE_K];
  __shared__ __align__(32) bf16_t lsB[2][TILE_N * TILE_K];

  const int tid  = threadIdx.x;
  const int lane = tid & 31;
  const int wid  = tid >> 5;
  const int wm   = wid >> 2;          // 0..1 -> 64-row slice
  const int wn   = wid & 3;           // 0..3 -> 32-col slice
  const int bM   = blockIdx.y * TILE_M;
  const int bN   = blockIdx.x * TILE_N;
  const int b    = blockIdx.z;

  const bf16_t* Ab = A  + (size_t)b * strideA;
  const bf16_t* Bb = Bt + (size_t)b * strideB;
  const int nk = K / TILE_K;

  // Edge blocks: zero the tile buffers once; async loads never touch the
  // out-of-range rows afterwards, so WMMA consumes zeros there.
  if (bM + TILE_M > M || bN + TILE_N > N) {
    uint4 z; z.x = z.y = z.z = z.w = 0u;
    for (int i = tid; i < 2 * TILE_M * TILE_K / 8; i += GEMM_THREADS)
      ((uint4*)&lsA[0][0])[i] = z;
    for (int i = tid; i < 2 * TILE_N * TILE_K / 8; i += GEMM_THREADS)
      ((uint4*)&lsB[0][0])[i] = z;
    __syncthreads();
  }

  v8f acc[4][2];
#pragma unroll
  for (int mi = 0; mi < 4; ++mi)
#pragma unroll
    for (int ni = 0; ni < 2; ++ni)
#pragma unroll
      for (int r = 0; r < 8; ++r) acc[mi][ni][r] = 0.0f;

  const int ml = lane & 15;           // M index (A) / N index (B) within 16
  const int h  = lane >> 4;           // lane-half -> K half (K = 16h .. 16h+15, contiguous)

  load_tiles_async(lsA[0], lsB[0], Ab, Bb, bM, bN, M, N, K, 0, tid);
  wait_async();
  __syncthreads();

  for (int kt = 0; kt < nk; ++kt) {
    const int cur = kt & 1;
    if (kt + 1 < nk)
      load_tiles_async(lsA[cur ^ 1], lsB[cur ^ 1], Ab, Bb, bM, bN, M, N, K,
                       kt + 1, tid);

    // B fragments: lane (n, h) reads K = h*16 .. h*16+15 contiguously (32 B).
    v16bf bf[2];
#pragma unroll
    for (int ni = 0; ni < 2; ++ni) {
      const bf16_t* r = &lsB[cur][(wn * 32 + ni * 16 + ml) * TILE_K + h * 16];
      bf[ni] = *(const v16bf*)r;
    }
    // A fragments: lane (m, h) reads K = h*16 .. h*16+15 contiguously.
    v16bf af[4];
#pragma unroll
    for (int mi = 0; mi < 4; ++mi) {
      const bf16_t* r = &lsA[cur][(wm * 64 + mi * 16 + ml) * TILE_K];
      FragA f;
      f.u[0] = *(const uint4*)(r + h * 16);
      f.u[1] = *(const uint4*)(r + h * 16 + 8);
      af[mi] = f.v;
    }
#pragma unroll
    for (int mi = 0; mi < 4; ++mi)
#pragma unroll
      for (int ni = 0; ni < 2; ++ni)
        acc[mi][ni] = __builtin_amdgcn_wmma_f32_16x16x32_bf16(
            false, af[mi], false, bf[ni], (short)0, acc[mi][ni], false, false);

    wait_async();     // next-stage async copies have landed in LDS
    __syncthreads();  // everyone done reading current stage
  }

  // ---- epilogue ----
  const float addv = addend ? addend[0] : 0.0f;
  const size_t obase = (size_t)b * (size_t)M * (size_t)N;
  const int m0 = h * 8;               // C/D layout: vgpr r -> M = h*8 + r

#pragma unroll
  for (int mi = 0; mi < 4; ++mi) {
#pragma unroll
    for (int ni = 0; ni < 2; ++ni) {
      const int gn = bN + wn * 32 + ni * 16 + ml;
      const int gmb = bM + wm * 64 + mi * 16 + m0;
#pragma unroll
      for (int r = 0; r < 8; ++r) {
        const int gm = gmb + r;
        if (gm < M && gn < N) {
          float v = acc[mi][ni][r] * alpha + addv;
          if (bias) v += bias[gn];
          if (act == 1)      v = (v > 0.0f) ? v : 0.01f * v;
          else if (act == 2) v = 0.5f * v * (1.0f + erff(v * 0.70710678118f));
          if (outF)  outF [obase + (size_t)gm * N + gn] = v;
          if (outB)  outB [obase + (size_t)gm * N + gn] = f2bf(v);
          if (outBT) outBT[obase + (size_t)gn * M + gm] = f2bf(v);
        }
      }
    }
  }
}

// ---------------------------------------------------------------------------
// fp32 [batch][R][C] -> bf16 [batch][C][R] (transpose-convert)
// ---------------------------------------------------------------------------
__global__ void tconv_kernel(const float* __restrict__ in, bf16_t* __restrict__ out,
                             int R, int C, size_t total)
{
  size_t idx = (size_t)blockIdx.x * blockDim.x + threadIdx.x;
  if (idx >= total) return;
  size_t rc  = (size_t)R * C;
  size_t bi  = idx / rc;
  size_t rem = idx - bi * rc;
  int r = (int)(rem / C);
  int c = (int)(rem - (size_t)r * C);
  out[bi * rc + (size_t)c * R + r] = f2bf(in[idx]);
}

// fp32 -> bf16, same layout
__global__ void conv_kernel(const float* __restrict__ in, bf16_t* __restrict__ out,
                            size_t total)
{
  size_t idx = (size_t)blockIdx.x * blockDim.x + threadIdx.x;
  if (idx < total) out[idx] = f2bf(in[idx]);
}

// ---------------------------------------------------------------------------
// LayerNorm over last dim D==1024; one 256-thread block per row.
// Writes bf16 (next WMMA GEMM's A operand) and optionally fp32 (in-place OK).
// ---------------------------------------------------------------------------
__global__ __launch_bounds__(256)
void layernorm_kernel(const float* __restrict__ in,
                      const float* __restrict__ gamma,
                      const float* __restrict__ beta,
                      float* __restrict__ outF, bf16_t* __restrict__ outB)
{
  const int row = blockIdx.x, tid = threadIdx.x;
  const float* x = in + (size_t)row * 1024;
  __shared__ float red[256];

  float v[4]; float s = 0.0f;
#pragma unroll
  for (int i = 0; i < 4; ++i) { v[i] = x[tid + i * 256]; s += v[i]; }
  red[tid] = s; __syncthreads();
  for (int o = 128; o > 0; o >>= 1) { if (tid < o) red[tid] += red[tid + o]; __syncthreads(); }
  const float mean = red[0] * (1.0f / 1024.0f);
  __syncthreads();

  float ss = 0.0f;
#pragma unroll
  for (int i = 0; i < 4; ++i) { float d = v[i] - mean; ss += d * d; }
  red[tid] = ss; __syncthreads();
  for (int o = 128; o > 0; o >>= 1) { if (tid < o) red[tid] += red[tid + o]; __syncthreads(); }
  const float rstd = rsqrtf(red[0] * (1.0f / 1024.0f) + 1e-5f);

#pragma unroll
  for (int i = 0; i < 4; ++i) {
    int d = tid + i * 256;
    float y = (v[i] - mean) * rstd * gamma[d] + beta[d];
    if (outF) outF[(size_t)row * 1024 + d] = y;
    if (outB) outB[(size_t)row * 1024 + d] = f2bf(y);
  }
}

// ---------------------------------------------------------------------------
// Softmax over rows of width 512 (scores fp32 -> attn_w bf16).
// ---------------------------------------------------------------------------
__global__ __launch_bounds__(256)
void softmax512_kernel(const float* __restrict__ in, bf16_t* __restrict__ out)
{
  const int row = blockIdx.x, tid = threadIdx.x;
  const float* x = in + (size_t)row * 512;
  __shared__ float red[256];

  float a = x[tid], b = x[tid + 256];
  float m = fmaxf(a, b);
  red[tid] = m; __syncthreads();
  for (int o = 128; o > 0; o >>= 1) { if (tid < o) red[tid] = fmaxf(red[tid], red[tid + o]); __syncthreads(); }
  const float mx = red[0]; __syncthreads();

  float ea = expf(a - mx), eb = expf(b - mx);
  red[tid] = ea + eb; __syncthreads();
  for (int o = 128; o > 0; o >>= 1) { if (tid < o) red[tid] += red[tid + o]; __syncthreads(); }
  const float inv = 1.0f / red[0];

  out[(size_t)row * 512 + tid]       = f2bf(ea * inv);
  out[(size_t)row * 512 + tid + 256] = f2bf(eb * inv);
}

// ---------------------------------------------------------------------------
// Attention pooling: scores = h @ Wp2 + bp2, softmax over N=512,
// pooled[d] = sum_n w[n] * fused_ln[b,n,d].  One block per batch.
// ---------------------------------------------------------------------------
__global__ __launch_bounds__(256)
void pool_kernel(const bf16_t* __restrict__ h,      // [B][512][64]
                 const float*  __restrict__ Wp2,    // [64]
                 const float*  __restrict__ bp2,    // [1]
                 const float*  __restrict__ fused,  // [B][512][1024] (normalized)
                 bf16_t* __restrict__ pooled)       // [B][1024]
{
  const int b = blockIdx.x, tid = threadIdx.x;
  __shared__ float sc[512];
  __shared__ float red[256];

  for (int n = tid; n < 512; n += 256) {
    const bf16_t* hr = h + ((size_t)b * 512 + n) * 64;
    float acc = 0.0f;
#pragma unroll
    for (int j = 0; j < 64; ++j) acc += bf2f(hr[j]) * Wp2[j];
    sc[n] = acc + bp2[0];
  }
  __syncthreads();

  float m = -3.0e38f;
  for (int n = tid; n < 512; n += 256) m = fmaxf(m, sc[n]);
  red[tid] = m; __syncthreads();
  for (int o = 128; o > 0; o >>= 1) { if (tid < o) red[tid] = fmaxf(red[tid], red[tid + o]); __syncthreads(); }
  const float mx = red[0]; __syncthreads();

  float s = 0.0f;
  for (int n = tid; n < 512; n += 256) { float e = expf(sc[n] - mx); sc[n] = e; s += e; }
  __syncthreads();
  red[tid] = s; __syncthreads();
  for (int o = 128; o > 0; o >>= 1) { if (tid < o) red[tid] += red[tid + o]; __syncthreads(); }
  const float inv = 1.0f / red[0];
  __syncthreads();

  for (int d = tid; d < 1024; d += 256) {
    float acc = 0.0f;
    const float* fb = fused + (size_t)b * 512 * 1024 + d;
    for (int n = 0; n < 512; ++n) acc += sc[n] * fb[(size_t)n * 1024];
    pooled[(size_t)b * 1024 + d] = f2bf(acc * inv);
  }
}

// rul[b] = | dot(h2[b], Wf3) + bf3 | ; one block per batch row.
__global__ __launch_bounds__(256)
void head_kernel(const bf16_t* __restrict__ h2, const float* __restrict__ Wf3,
                 const float* __restrict__ bf3, float* __restrict__ out)
{
  const int b = blockIdx.x, tid = threadIdx.x;
  __shared__ float red[256];
  float acc = 0.0f;
  for (int j = tid; j < 2048; j += 256) acc += bf2f(h2[(size_t)b * 2048 + j]) * Wf3[j];
  red[tid] = acc; __syncthreads();
  for (int o = 128; o > 0; o >>= 1) { if (tid < o) red[tid] += red[tid + o]; __syncthreads(); }
  if (tid == 0) out[b] = fabsf(red[0] + bf3[0]);
}

// ---------------------------------------------------------------------------
// Host-side orchestration
// ---------------------------------------------------------------------------
static inline size_t align_up(size_t x) { return (x + 255) & ~(size_t)255; }

extern "C" void kernel_launch(void* const* d_in, const int* in_sizes, int n_in,
                              void* d_out, int out_size, void* d_ws, size_t ws_size,
                              hipStream_t stream)
{
  (void)in_sizes; (void)n_in; (void)out_size; (void)ws_size;
  const int B = 128, T = 1024, C = 512, D = 1024, DF = 2048, H = 64;
  const float inv_sqrt_d = 0.03125f;   // 1/sqrt(1024)

  const float* x_enc = (const float*)d_in[0];
  // d_in[1] (aging) is unused by the reference
  const float* W_emb = (const float*)d_in[2];  const float* b_emb = (const float*)d_in[3];
  const float* g_s   = (const float*)d_in[4];  const float* b_s   = (const float*)d_in[5];
  const float* basis = (const float*)d_in[6];
  const float* Wq    = (const float*)d_in[7];  const float* bq    = (const float*)d_in[8];
  const float* Wk    = (const float*)d_in[9];  const float* bk    = (const float*)d_in[10];
  const float* Wv    = (const float*)d_in[11]; const float* bv    = (const float*)d_in[12];
  const float* age_scale = (const float*)d_in[13];
  const float* g_f   = (const float*)d_in[14]; const float* b_f   = (const float*)d_in[15];
  const float* Wp1   = (const float*)d_in[16]; const float* bp1   = (const float*)d_in[17];
  const float* Wp2   = (const float*)d_in[18]; const float* bp2   = (const float*)d_in[19];
  const float* Wf1   = (const float*)d_in[20]; const float* bf1   = (const float*)d_in[21];
  const float* Wf2   = (const float*)d_in[22]; const float* bf2   = (const float*)d_in[23];
  const float* Wf3   = (const float*)d_in[24]; const float* bf3   = (const float*)d_in[25];

  char* w = (char*)d_ws;
  size_t off = 0;
  auto alloc = [&](size_t bytes) { size_t o = off; off = align_up(off + bytes); return o; };

  // Liveness-aliased workspace layout (~1.0 GB total).
  const size_t o_xT    = alloc((size_t)B * C * T * 2);   // x_enc^T bf16; reused as Q bf16
  const size_t o_We    = alloc((size_t)T * D * 2);       // W_emb^T bf16
  const size_t o_big   = alloc((size_t)B * C * D * 4);   // sensor f32 -> scores f32 -> fused f32
  const size_t o_bigb  = alloc((size_t)B * C * D * 2);   // sensor bf16 -> fused_ln bf16
  const size_t o_WqT   = alloc((size_t)D * D * 2);
  const size_t o_WkT   = alloc((size_t)D * D * 2);
  const size_t o_WvT   = alloc((size_t)D * D * 2);
  const size_t o_bas   = alloc((size_t)C * D * 2);       // basis bf16 (already [N][K])
  const size_t o_Wp1T  = alloc((size_t)D * H * 2);
  const size_t o_Wf1T  = alloc((size_t)D * DF * 2);
  const size_t o_Wf2T  = alloc((size_t)DF * DF * 2);
  const size_t o_K     = alloc((size_t)B * C * D * 2);   // K bf16 (row-major == Bt for QK^T)
  const size_t o_VT    = alloc((size_t)B * C * D * 2);   // V^T bf16 [D][C]
  const size_t o_ageT  = alloc((size_t)B * C * C * 2);   // age_bias^T bf16
  const size_t o_scT   = alloc((size_t)B * C * C * 2);   // scale^T bf16
  const size_t o_aw    = alloc((size_t)B * C * C * 2);   // attn_w bf16
  const size_t o_h     = alloc((size_t)B * C * H * 2);   // pooling hidden bf16
  const size_t o_pool  = alloc((size_t)B * D * 2);       // pooled bf16
  const size_t o_h1    = alloc((size_t)B * DF * 2);
  const size_t o_h2    = alloc((size_t)B * DF * 2);

  const size_t o_Q = o_xT;   // x_enc^T is dead after GEMM1

  auto bw = [&](size_t o) { return (bf16_t*)(w + o); };

  auto gemm = [&](const bf16_t* A, size_t sA, const bf16_t* Bt, size_t sB,
                  const float* bias, const float* addp, float alpha, int act,
                  float* oF, bf16_t* oB, bf16_t* oBT,
                  int batch, int M, int N, int Kd) {
    dim3 grid((N + TILE_N - 1) / TILE_N, (M + TILE_M - 1) / TILE_M, batch);
    gemm_bf16_wmma<<<grid, GEMM_THREADS, 0, stream>>>(
        A, sA, Bt, sB, bias, addp, alpha, act, oF, oB, oBT, M, N, Kd);
  };
  auto tconv = [&](const float* in, bf16_t* outp, int R, int Cc, size_t total) {
    tconv_kernel<<<(unsigned)((total + 255) / 256), 256, 0, stream>>>(in, outp, R, Cc, total);
  };

  // ---- weight / input conversions (bf16, B-operands pre-transposed to [N][K]) ----
  tconv(x_enc, bw(o_xT),  T, C,  (size_t)B * T * C);       // [B][C][T]
  tconv(W_emb, bw(o_We),  T, D,  (size_t)T * D);           // [D][T]
  tconv(Wq,    bw(o_WqT), D, D,  (size_t)D * D);
  tconv(Wk,    bw(o_WkT), D, D,  (size_t)D * D);
  tconv(Wv,    bw(o_WvT), D, D,  (size_t)D * D);
  tconv(Wp1,   bw(o_Wp1T), D, H, (size_t)D * H);
  tconv(Wf1,   bw(o_Wf1T), D, DF, (size_t)D * DF);
  tconv(Wf2,   bw(o_Wf2T), DF, DF, (size_t)DF * DF);
  conv_kernel<<<(unsigned)(((size_t)C * D + 255) / 256), 256, 0, stream>>>(
      basis, bw(o_bas), (size_t)C * D);

  // ---- 1) sensor = x_enc^T @ W_emb + b_emb  (fp32 out) ----
  gemm(bw(o_xT), (size_t)C * T, bw(o_We), 0, b_emb, nullptr, 1.0f, 0,
       (float*)(w + o_big), nullptr, nullptr, B, C, D, T);

  // ---- 2) LayerNorm(sensor) -> bf16 ----
  layernorm_kernel<<<B * C, 256, 0, stream>>>((float*)(w + o_big), g_s, b_s,
                                              nullptr, bw(o_bigb));

  // ---- 3) Q, K, V ----
  gemm(bw(o_bigb), (size_t)C * D, bw(o_WqT), 0, bq, nullptr, 1.0f, 0,
       nullptr, bw(o_Q), nullptr, B, C, D, D);
  gemm(bw(o_bigb), (size_t)C * D, bw(o_WkT), 0, bk, nullptr, 1.0f, 0,
       nullptr, bw(o_K), nullptr, B, C, D, D);
  gemm(bw(o_bigb), (size_t)C * D, bw(o_WvT), 0, bv, nullptr, 1.0f, 0,
       nullptr, nullptr, bw(o_VT), B, C, D, D);             // V stored transposed

  // ---- 4) scale^T = (Q K^T * inv_sqrt_d + age_scale)^T ;  age_bias^T ----
  gemm(bw(o_Q), (size_t)C * D, bw(o_K), (size_t)C * D, nullptr, age_scale,
       inv_sqrt_d, 0, nullptr, nullptr, bw(o_scT), B, C, C, D);
  gemm(bw(o_Q), (size_t)C * D, bw(o_bas), 0, nullptr, nullptr,
       inv_sqrt_d, 0, nullptr, nullptr, bw(o_ageT), B, C, C, D);

  // ---- 5) scores = age_bias^T @ scale  (fp32, reuses the big buffer) ----
  gemm(bw(o_ageT), (size_t)C * C, bw(o_scT), (size_t)C * C, nullptr, nullptr,
       1.0f, 0, (float*)(w + o_big), nullptr, nullptr, B, C, C, C);

  // ---- 6) softmax -> attn_w bf16 ----
  softmax512_kernel<<<B * C, 256, 0, stream>>>((float*)(w + o_big), bw(o_aw));

  // ---- 7) fused = attn_w @ V * inv_sqrt_d (fp32, overwrites scores) ----
  gemm(bw(o_aw), (size_t)C * C, bw(o_VT), (size_t)C * D, nullptr, nullptr,
       inv_sqrt_d, 0, (float*)(w + o_big), nullptr, nullptr, B, C, D, C);

  // ---- 8) LayerNorm(fused) -> fp32 in place + bf16 ----
  layernorm_kernel<<<B * C, 256, 0, stream>>>((float*)(w + o_big), g_f, b_f,
                                              (float*)(w + o_big), bw(o_bigb));

  // ---- 9) pooling hidden h = GELU(fused_ln @ Wp1 + bp1) ----
  gemm(bw(o_bigb), (size_t)C * D, bw(o_Wp1T), 0, bp1, nullptr, 1.0f, 2,
       nullptr, bw(o_h), nullptr, B, C, H, D);

  // ---- 10) attention pooling -> pooled bf16 [B][D] ----
  pool_kernel<<<B, 256, 0, stream>>>(bw(o_h), Wp2, bp2, (float*)(w + o_big),
                                     bw(o_pool));

  // ---- 11) fc head ----
  gemm(bw(o_pool), 0, bw(o_Wf1T), 0, bf1, nullptr, 1.0f, 1,
       nullptr, bw(o_h1), nullptr, 1, B, DF, D);
  gemm(bw(o_h1), 0, bw(o_Wf2T), 0, bf2, nullptr, 1.0f, 0,
       nullptr, bw(o_h2), nullptr, 1, B, DF, DF);
  head_kernel<<<B, 256, 0, stream>>>(bw(o_h2), Wf3, bf3, (float*)d_out);
}
